// idct_layer_hy_12996571038116
// MI455X (gfx1250) — compile-verified, matrix operand order checked by reading, not run on hardware
//
#include <hip/hip_runtime.h>

// CDNA5 wave32 WMMA types
typedef _Float16 v16h __attribute__((ext_vector_type(16)));
typedef _Float16 v8h  __attribute__((ext_vector_type(8)));
typedef _Float16 v4h  __attribute__((ext_vector_type(4)));
typedef float    v8f  __attribute__((ext_vector_type(8)));

#define N_ROWS 1024   // n: transform length (= M = K of the GEMM)
#define N_COLS 512    // m: columns (= N of the GEMM)
#define N_BATCH 32
#define BM 128        // workgroup M tile
#define BN 256        // workgroup N tile
#define BK 32         // K step (matches 16x16x32 WMMA)
#define LDSS 48       // padded LDS row stride in halves (96B: 16B-aligned rows)

// ---------------------------------------------------------------------------
// Kernel 1: build the orthonormal DCT-III matrix C[m][k] in f16.
//   C[m][0] = 1/sqrt(n) = 1/32 (exact)
//   C[m][k] = sqrt(2/n) * cos(pi*(2m+1)*k/(2n)),  k >= 1
// Phase reduced mod 4n = 4096 in integers => cosf argument < 2*pi (accurate).
// ---------------------------------------------------------------------------
__global__ void build_dct3_matrix(_Float16* __restrict__ C) {
    int idx = blockIdx.x * blockDim.x + threadIdx.x;  // 0 .. 1024*1024-1
    int m = idx >> 10;
    int k = idx & 1023;
    float val;
    if (k == 0) {
        val = 0.03125f;                                // 1/sqrt(1024)
    } else {
        int t = ((2 * m + 1) * k) & 4095;              // phase mod 4n
        float ang = (float)t * 1.5339807878856412e-3f; // * pi/2048
        val = 0.04419417382415922f * cosf(ang);        // sqrt(2/1024)
    }
    C[idx] = (_Float16)val;
}

// ---------------------------------------------------------------------------
// Kernel 2: Out[b] = C (1024x1024, f16) * X[b] (1024x512, f32->f16), f32 accum.
// 256 threads = 8 wave32s, waves tiled 2(M) x 4(N); each wave owns 64x64
// (16 WMMA tiles, 128 accumulator VGPRs). Global->register prefetch is
// software-pipelined across the LDS barrier.
// ---------------------------------------------------------------------------
__global__ void __launch_bounds__(256)
dct3_gemm(const float* __restrict__ X, const _Float16* __restrict__ C,
          float* __restrict__ Y) {
    __shared__ _Float16 sA[BM * LDSS];   // A tile: [row][k]            12 KB
    __shared__ _Float16 sB[BN * LDSS];   // B tile: [col][k] transposed 24 KB

    const int tid  = threadIdx.x;
    const int lane = tid & 31;
    const int wid  = tid >> 5;
    const int wm   = wid & 1;            // wave row group: rows wm*64..+63
    const int wn   = wid >> 1;           // wave col group: cols wn*64..+63
    const int lr   = lane & 15;
    const int kh   = (lane < 16) ? 0 : 8;  // ISA 16-bit operand K-half select

    const int rowBase = blockIdx.y * BM;
    const int colBase = blockIdx.x * BN;
    const size_t batchOff = (size_t)blockIdx.z * (size_t)(N_ROWS * N_COLS);

    // --- staging decompositions (pure functions of tid) ---
    // A: 128x32 halves; thread-task p in {0,1}: 8 contiguous halves (16B)
    const int aE = tid * 8;
    const int aR0 = aE >> 5, aC0 = aE & 31;          // pass 0
    const int aR1 = (aE + 2048) >> 5, aC1 = aC0;     // pass 1 (+256*8 halves)
    // B: 32k x 256col f32; task q: 4 k-rows x 4 cols
    const int q0  = tid;            // pass 0: kg 0..3
    const int q1  = tid + 256;      // pass 1: kg 4..7
    const int kg0 = q0 >> 6, cg0 = (q0 & 63) << 2;
    const int kg1 = q1 >> 6, cg1 = (q1 & 63) << 2;

    v8f acc[4][4];
#pragma unroll
    for (int i = 0; i < 4; ++i)
#pragma unroll
        for (int j = 0; j < 4; ++j) acc[i][j] = (v8f)(0.0f);

    // prefetch registers
    v8h pa[2];
    float4 pb[8];

    auto loadA = [&](int k0) {
        pa[0] = *(const v8h*)(C + (size_t)(rowBase + aR0) * N_ROWS + k0 + aC0);
        pa[1] = *(const v8h*)(C + (size_t)(rowBase + aR1) * N_ROWS + k0 + aC1);
    };
    auto loadB = [&](int k0) {
#pragma unroll
        for (int r = 0; r < 4; ++r) {
            pb[r] = *(const float4*)(X + batchOff +
                     (size_t)(k0 + kg0 * 4 + r) * N_COLS + colBase + cg0);
            pb[4 + r] = *(const float4*)(X + batchOff +
                     (size_t)(k0 + kg1 * 4 + r) * N_COLS + colBase + cg1);
        }
    };
    auto stageLDS = [&]() {
        *(v8h*)(&sA[aR0 * LDSS + aC0]) = pa[0];
        *(v8h*)(&sA[aR1 * LDSS + aC1]) = pa[1];
#pragma unroll
        for (int p = 0; p < 2; ++p) {
            const int kg = p ? kg1 : kg0;
            const int cg = p ? cg1 : cg0;
            const float* f0 = (const float*)&pb[p * 4 + 0];
            const float* f1 = (const float*)&pb[p * 4 + 1];
            const float* f2 = (const float*)&pb[p * 4 + 2];
            const float* f3 = (const float*)&pb[p * 4 + 3];
#pragma unroll
            for (int j = 0; j < 4; ++j) {   // pack 4 K values -> one b64 store
                v4h h;
                h[0] = (_Float16)f0[j];
                h[1] = (_Float16)f1[j];
                h[2] = (_Float16)f2[j];
                h[3] = (_Float16)f3[j];
                *(v4h*)(&sB[(cg + j) * LDSS + kg * 4]) = h;
            }
        }
    };

    loadA(0);
    loadB(0);

    for (int kt = 0; kt < N_ROWS / BK; ++kt) {
        stageLDS();
        __syncthreads();

        // issue next tile's global loads while WMMAs consume this LDS tile
        if (kt + 1 < N_ROWS / BK) {
            loadA((kt + 1) * BK);
            loadB((kt + 1) * BK);
        }

        // ---- fragments per ISA 16-bit A/B layout:
        // lanes 0-15: K 0-7 then 16-23; lanes 16-31: K 8-15 then 24-31 ----
        v16h afrag[4];
#pragma unroll
        for (int i = 0; i < 4; ++i) {
            int row = wm * 64 + i * 16 + lr;
            v8h lo = *(const v8h*)(&sA[row * LDSS + kh]);
            v8h hi = *(const v8h*)(&sA[row * LDSS + 16 + kh]);
#pragma unroll
            for (int e = 0; e < 8; ++e) { afrag[i][e] = lo[e]; afrag[i][e + 8] = hi[e]; }
        }
#pragma unroll
        for (int j = 0; j < 4; ++j) {
            int col = wn * 64 + j * 16 + lr;
            v8h lo = *(const v8h*)(&sB[col * LDSS + kh]);
            v8h hi = *(const v8h*)(&sB[col * LDSS + 16 + kh]);
            v16h bfrag;
#pragma unroll
            for (int e = 0; e < 8; ++e) { bfrag[e] = lo[e]; bfrag[e + 8] = hi[e]; }
#pragma unroll
            for (int i = 0; i < 4; ++i)
                acc[i][j] = __builtin_amdgcn_wmma_f32_16x16x32_f16(
                    false, afrag[i], false, bfrag,
                    (short)0, acc[i][j], false, false);
        }

        __syncthreads();
    }

    // ---- Store D per ISA 16x16 f32 C/D layout:
    // VGPR v: lanes 0-15 -> M=v, N=lane; lanes 16-31 -> M=8+v, N=lane-16 ----
#pragma unroll
    for (int i = 0; i < 4; ++i) {
#pragma unroll
        for (int j = 0; j < 4; ++j) {
            int r0  = rowBase + wm * 64 + i * 16 + ((lane < 16) ? 0 : 8);
            int col = colBase + wn * 64 + j * 16 + lr;
#pragma unroll
            for (int v = 0; v < 8; ++v)
                Y[batchOff + (size_t)(r0 + v) * N_COLS + col] = acc[i][j][v];
        }
    }
}

extern "C" void kernel_launch(void* const* d_in, const int* in_sizes, int n_in,
                              void* d_out, int out_size, void* d_ws, size_t ws_size,
                              hipStream_t stream) {
    (void)in_sizes; (void)n_in; (void)out_size; (void)ws_size;
    const float* x   = (const float*)d_in[0];
    float*       out = (float*)d_out;
    _Float16*    Cws = (_Float16*)d_ws;   // 1024*1024 f16 = 2 MiB of scratch

    // 1) Build the DCT-III cosine matrix (rebuilt every call: deterministic).
    build_dct3_matrix<<<(N_ROWS * N_ROWS) / 256, 256, 0, stream>>>(Cws);

    // 2) Batched GEMM: Out[b] = C * X[b].
    dim3 grid(N_COLS / BN, N_ROWS / BM, N_BATCH);
    dct3_gemm<<<grid, 256, 0, stream>>>(x, Cws, out);
}
